// Net_23210003267823
// MI455X (gfx1250) — matrix operand order repaired and checked by reading, not compiled, not run on hardware
//
#include <hip/hip_runtime.h>
#include <hip/hip_bf16.h>

// ---------------------------------------------------------------------------
// BiGCN rumor-net forward for MI455X (gfx1250, wave32, WMMA).
// Dominant GEMM: H = x(20000x5000 f32->bf16) @ [W1_td|W1_bu](5000x128 bf16)
// via v_wmma_f32_16x16x32_bf16.  HBM-bound (400MB of x @ 23.3TB/s ~= 17us);
// all intermediates (<35MB) are L2-resident (192MB), so edge aggregation
// uses f32 atomics at L2 speed.  conv2's 5000-K term is collapsed to a
// 128x5000x128 GEMM over the G distinct root rows.
// GEMM inner loop is software-pipelined: A tile kt+1 is prefetched to
// registers (non-temporal) behind the WMMA chain of tile kt, and all four
// B fragments are loaded before the WMMA chain so waits can be partial.
// ---------------------------------------------------------------------------

#define NN   20000
#define EE   320000
#define GG   128
#define IN_F 5000
#define KTILES 157      // ceil(5000/32)
#define LDC  128

typedef __attribute__((ext_vector_type(16))) __bf16 bf16x16;
typedef __attribute__((ext_vector_type(8)))  float  f32x8;
typedef __attribute__((ext_vector_type(4)))  float  f32x4;   // native vector for NT loads

__device__ __forceinline__ unsigned bf16rne(float x) {
    unsigned u = __float_as_uint(x);
    return (u + 0x7FFFu + ((u >> 16) & 1u)) >> 16;   // round-to-nearest-even
}
__device__ __forceinline__ unsigned pack2bf(float lo, float hi) {
    return bf16rne(lo) | (bf16rne(hi) << 16);
}

// ---------------------------------------------------------------------------
// Pack two (5000 x 64) fp32 weight blocks into WMMA B-fragment order (bf16).
// Fragment (kt,nt): 256 dwords; lane l holds dwords [l*8 .. l*8+7] where
// VGPR v = K pair (2v,2v+1), lanes 16..31 shifted by K+16 (contiguous K order, same as the A fragment).
// ---------------------------------------------------------------------------
__global__ __launch_bounds__(256)
void k_pack(const float* __restrict__ Wtd, const float* __restrict__ Wbu,
            int rowoff, unsigned* __restrict__ out) {
    int t = blockIdx.x * 256 + threadIdx.x;        // < 157*8*256
    int fi = t >> 8, r = t & 255;
    int lane = r >> 3, v = r & 7;
    int kt = fi >> 3, nt = fi & 7;
    int k = kt * 32 + ((lane & 16) ? 16 : 0) + 2 * v;
    int n = nt * 16 + (lane & 15);
    const float* W = (n < 64) ? Wtd : Wbu;
    int nn = n & 63;
    float lo = (k     < IN_F) ? W[(long)(k + rowoff) * 64 + nn]     : 0.f;
    float hi = (k + 1 < IN_F) ? W[(long)(k + 1 + rowoff) * 64 + nn] : 0.f;
    out[t] = pack2bf(lo, hi);
}

// ---------------------------------------------------------------------------
// C(Mx128) = A(Mx5000 f32, optional row-gather + relu) @ Bpack (bf16 frags).
// Block: 256 thr = 8 waves; tile 64(M) x 128(N); K step 32.
// Wave w: rows (w&3)*16..+15, cols (w>>2)*64..+63 (4 fragments).
// ---------------------------------------------------------------------------
template <bool GR>
__global__ __launch_bounds__(256)
void k_gemm(const float* __restrict__ A, int M,
            const unsigned* __restrict__ Bp, const int* __restrict__ rowidx,
            float* __restrict__ C) {
    __shared__ unsigned ldsA[64 * 20];             // 64 rows x 32 K bf16, stride 20 dw
    const int tid  = threadIdx.x;
    const int lane = tid & 31;
    const int w    = tid >> 5;
    const int mblk = blockIdx.x * 64;

    // staging role: thread -> (row 0..63, quarter 0..3 of the 32-K strip)
    const int srow = tid >> 2;
    const int sq   = tid & 3;
    const int gr   = mblk + srow;
    const bool rvalid = gr < M;
    long arow = 0;
    if (rvalid) arow = (long)(GR ? rowidx[gr] : gr) * IN_F;

    // A-fragment LDS address (contiguous K: lane half h holds K 16h..16h+15, matching Bpack)
    const int adw0 = ((w & 3) * 16 + (lane & 15)) * 20 + ((lane & 16) ? 8 : 0);
    const int ntb  = (w >> 2) * 4;

    f32x8 acc[4] = {f32x8{}, f32x8{}, f32x8{}, f32x8{}};

    float f[8];                                     // current A tile (registers)
    auto load_tile = [&](int kt) {
        const int k0 = kt * 32 + sq * 8;
        if (rvalid && (k0 + 8 <= IN_F)) {
            f32x4 p0 = __builtin_nontemporal_load((const f32x4*)(A + arow + k0));
            f32x4 p1 = __builtin_nontemporal_load((const f32x4*)(A + arow + k0 + 4));
            f[0] = p0[0]; f[1] = p0[1]; f[2] = p0[2]; f[3] = p0[3];
            f[4] = p1[0]; f[5] = p1[1]; f[6] = p1[2]; f[7] = p1[3];
        } else {
#pragma unroll
            for (int j = 0; j < 8; ++j)
                f[j] = (rvalid && (k0 + j) < IN_F) ? A[arow + k0 + j] : 0.f;
        }
        if (GR) {
#pragma unroll
            for (int j = 0; j < 8; ++j) f[j] = fmaxf(f[j], 0.f);
        }
    };

    load_tile(0);                                   // prologue

    for (int kt = 0; kt < KTILES; ++kt) {
        // convert the tile we already hold, publish to LDS
        uint4 pk;
        pk.x = pack2bf(f[0], f[1]);
        pk.y = pack2bf(f[2], f[3]);
        pk.z = pack2bf(f[4], f[5]);
        pk.w = pack2bf(f[6], f[7]);
        __syncthreads();                            // prior iteration's reads done
        *(uint4*)&ldsA[srow * 20 + sq * 4] = pk;    // ds_store_b128
        __syncthreads();

        // all four B fragments first (waits can retire these while A
        // prefetch below stays outstanding)
        union BU { uint4 q[2]; bf16x16 v; } bu_[4];
#pragma unroll
        for (int fr = 0; fr < 4; ++fr) {
            const unsigned* bp = Bp + (((long)(kt * 8 + ntb + fr)) << 8) + lane * 8;
            bu_[fr].q[0] = *(const uint4*)bp;       // global_load_b128 x2
            bu_[fr].q[1] = *(const uint4*)(bp + 4);
        }

        // A fragment from LDS
        union AU { uint4 q[2]; bf16x16 v; } au;
        au.q[0] = *(const uint4*)&ldsA[adw0];       // ds_load_b128 x2
        au.q[1] = *(const uint4*)&ldsA[adw0 + 4];

        // prefetch next A tile behind the WMMA chain (issued after B loads
        // so in-order load completion doesn't force waiting on it)
        if (kt + 1 < KTILES) load_tile(kt + 1);

#pragma unroll
        for (int fr = 0; fr < 4; ++fr) {
            acc[fr] = __builtin_amdgcn_wmma_f32_16x16x32_bf16(
                false, au.v, false, bu_[fr].v, (short)0, acc[fr], false, false);
        }
    }

    // epilogue: C/D 16x16 f32 layout -> row-major store
#pragma unroll
    for (int fr = 0; fr < 4; ++fr) {
        const int n  = (ntb + fr) * 16 + (lane & 15);
        const int m0 = mblk + (w & 3) * 16 + ((lane & 16) ? 8 : 0);
#pragma unroll
        for (int r = 0; r < 8; ++r) {
            int m = m0 + r;
            if (m < M) C[(long)m * LDC + n] = acc[fr][r];
        }
    }
}

// --------------------------- degree / norm ---------------------------------
__global__ __launch_bounds__(256)
void k_deg_init(float* __restrict__ deg) {
    int t = blockIdx.x * 256 + threadIdx.x;
    if (t < 2 * NN) deg[t] = 1.0f;                   // self-loop
}
__global__ __launch_bounds__(256)
void k_deg_count(const int* __restrict__ etd, const int* __restrict__ ebu,
                 float* __restrict__ deg) {
    int t = blockIdx.x * 256 + threadIdx.x;          // < 2*EE
    int b = (t >= EE) ? 1 : 0;
    int e = t - b * EE;
    const int* ei = b ? ebu : etd;
    atomicAdd(&deg[b * NN + ei[EE + e]], 1.0f);
}
__global__ __launch_bounds__(256)
void k_dinv(const float* __restrict__ deg, float* __restrict__ dinv) {
    int t = blockIdx.x * 256 + threadIdx.x;
    if (t < 2 * NN) dinv[t] = rsqrtf(deg[t]);
}

// ---------------- aggregation: self-loop + bias init, edge atomics ----------
__global__ __launch_bounds__(256)
void k_agg_init(const float* __restrict__ H, const float* __restrict__ dinv,
                const float* __restrict__ btd, const float* __restrict__ bbu,
                float* __restrict__ agg) {
    int t = blockIdx.x * 256 + threadIdx.x;          // < 2560000
    int i = t >> 7, c = t & 127;
    int b = c >> 6;
    float di = dinv[b * NN + i];
    float bias = (c < 64) ? btd[c] : bbu[c - 64];
    agg[t] = H[t] * di * di + bias;
}
__global__ __launch_bounds__(256)
void k_agg_edge(const int* __restrict__ etd, const int* __restrict__ ebu,
                const float* __restrict__ dinv, const float* __restrict__ H,
                float* __restrict__ agg) {
    int t = blockIdx.x * 256 + threadIdx.x;          // < 2*EE*64
    int c  = t & 63;
    int eb = t >> 6;
    int b  = (eb >= EE) ? 1 : 0;
    int e  = eb - b * EE;
    const int* ei = b ? ebu : etd;
    int src = ei[e], dst = ei[EE + e];
    float wgt = dinv[b * NN + src] * dinv[b * NN + dst];
    float v = H[(long)src * 128 + b * 64 + c] * wgt;
    atomicAdd(&agg[(long)dst * 128 + b * 64 + c], v);
}

// ---------- H2 = relu(h1)@W2[0:64] + R[batch]  (K=64, fp32, trivial) --------
__global__ __launch_bounds__(256)
void k_h2(const float* __restrict__ agg1, const float* __restrict__ W2td,
          const float* __restrict__ W2bu, const float* __restrict__ R,
          const int* __restrict__ batch, float* __restrict__ H) {
    __shared__ float srow[256];
    int t = blockIdx.x * 256 + threadIdx.x;          // < 2560000
    int i = t >> 7, c = t & 127;
    srow[threadIdx.x] = agg1[(long)i * 128 + c];
    __syncthreads();
    const int rb   = threadIdx.x & 128;              // row-in-block offset (0/128)
    const int boff = (c < 64) ? 0 : 64;
    const float* wcol = (c < 64) ? (W2td + c) : (W2bu + (c - 64));
    float acc = R[(long)batch[i] * 128 + c];
#pragma unroll 8
    for (int k = 0; k < 64; ++k)
        acc += fmaxf(srow[rb + boff + k], 0.f) * wcol[k * 64];
    H[t] = acc;
}

// --------------------------- pooling + head --------------------------------
__global__ __launch_bounds__(256)
void k_pool_init(float* __restrict__ pool, float* __restrict__ cnt) {
    int t = blockIdx.x * 256 + threadIdx.x;
    if (t < GG * 256) pool[t] = 0.f;
    else if (t < GG * 256 + GG) cnt[t - GG * 256] = 0.f;
}
__global__ __launch_bounds__(256)
void k_pool_acc(const float* __restrict__ agg2, const int* __restrict__ batch,
                float* __restrict__ pool, float* __restrict__ cnt) {
    int t = blockIdx.x * 256 + threadIdx.x;          // < 2560000
    int i = t >> 7, c = t & 127;
    int g = batch[i];
    float v = fmaxf(agg2[t], 0.f);                   // relu(conv2 out)
    int f = (c < 64) ? c : c + 64;                   // td->[0,64), bu->[128,192)
    atomicAdd(&pool[g * 256 + f], v);
    if (c == 0) atomicAdd(&cnt[g], 1.0f);
}
__global__ void k_final(const float* __restrict__ pool, const float* __restrict__ cnt,
                        const float* __restrict__ agg1, const int* __restrict__ rooti,
                        const float* __restrict__ fcW, const float* __restrict__ fcb,
                        float* __restrict__ out) {
    int g = threadIdx.x;
    if (g >= GG) return;
    float c = fmaxf(cnt[g], 1.0f);
    int ri = rooti[g];
    float l0 = fcb[0], l1 = fcb[1];
    for (int f = 0; f < 256; ++f) {
        float h;
        if (f < 64)       h = pool[g * 256 + f] / c;              // td conv2 mean
        else if (f < 128) h = agg1[(long)ri * 128 + (f - 64)];    // td root2 (mean == root row)
        else if (f < 192) h = pool[g * 256 + f] / c;              // bu conv2 mean
        else              h = agg1[(long)ri * 128 + 64 + (f - 192)]; // bu root2
        l0 += h * fcW[f * 2];
        l1 += h * fcW[f * 2 + 1];
    }
    float m = fmaxf(l0, l1);
    float lse = m + logf(expf(l0 - m) + expf(l1 - m));
    out[g * 2]     = l0 - lse;
    out[g * 2 + 1] = l1 - lse;
}

// ---------------------------------------------------------------------------
extern "C" void kernel_launch(void* const* d_in, const int* in_sizes, int n_in,
                              void* d_out, int out_size, void* d_ws, size_t ws_size,
                              hipStream_t stream) {
    const float* x     = (const float*)d_in[0];
    const int*   ei_td = (const int*)d_in[1];
    const int*   ei_bu = (const int*)d_in[2];
    const int*   batch = (const int*)d_in[3];
    const int*   rooti = (const int*)d_in[4];
    const float* W1td  = (const float*)d_in[5];
    const float* b1td  = (const float*)d_in[6];
    const float* W2td  = (const float*)d_in[7];
    const float* b2td  = (const float*)d_in[8];
    const float* W1bu  = (const float*)d_in[9];
    const float* b1bu  = (const float*)d_in[10];
    const float* W2bu  = (const float*)d_in[11];
    const float* b2bu  = (const float*)d_in[12];
    const float* fcW   = (const float*)d_in[13];
    const float* fcb   = (const float*)d_in[14];
    float* out = (float*)d_out;

    // workspace carve-up (4-byte units)
    unsigned* B1   = (unsigned*)d_ws;                  // 321536
    unsigned* B2   = B1 + 321536;                      // 321536
    float*    H    = (float*)(B2 + 321536);            // 2,560,000 (H1, reused as H2)
    float*    agg1 = H    + 2560000;                   // 2,560,000
    float*    agg2 = agg1 + 2560000;                   // 2,560,000
    float*    R    = agg2 + 2560000;                   // 16384
    float*    deg  = R    + 16384;                     // 40000
    float*    dinv = deg  + 40000;                     // 40000
    float*    pool = dinv + 40000;                     // 32768
    float*    cnt  = pool + 32768;                     // 128

    // weight packing (bf16 fragment layout)
    k_pack<<<1256, 256, 0, stream>>>(W1td, W1bu, 0,  B1);
    k_pack<<<1256, 256, 0, stream>>>(W2td, W2bu, 64, B2);   // W2 rows 64..5063

    // degrees / symmetric normalization (per branch)
    k_deg_init <<<157,  256, 0, stream>>>(deg);
    k_deg_count<<<2500, 256, 0, stream>>>(ei_td, ei_bu, deg);
    k_dinv     <<<157,  256, 0, stream>>>(deg, dinv);

    // big GEMM: H = x @ [W1_td | W1_bu]          (20000 x 5000 x 128)
    k_gemm<false><<<313, 256, 0, stream>>>(x, NN, B1, nullptr, H);
    // root GEMM: R = relu(x[root]) @ [W2b_td | W2b_bu]   (128 x 5000 x 128)
    k_gemm<true><<<2, 256, 0, stream>>>(x, GG, B2, rooti, R);

    // conv1 aggregation -> agg1 (= h1 per branch, pre-relu)
    k_agg_init<<<10000,  256, 0, stream>>>(H, dinv, b1td, b1bu, agg1);
    k_agg_edge<<<160000, 256, 0, stream>>>(ei_td, ei_bu, dinv, H, agg1);

    // conv2 pre-messages: H2 = relu(h1) @ W2[0:64] + R[batch]
    k_h2<<<10000, 256, 0, stream>>>(agg1, W2td, W2bu, R, batch, H);

    // conv2 aggregation -> agg2
    k_agg_init<<<10000,  256, 0, stream>>>(H, dinv, b2td, b2bu, agg2);
    k_agg_edge<<<160000, 256, 0, stream>>>(ei_td, ei_bu, dinv, H, agg2);

    // scatter-mean pooling + FC + log_softmax
    k_pool_init<<<129,   256, 0, stream>>>(pool, cnt);
    k_pool_acc <<<10000, 256, 0, stream>>>(agg2, batch, pool, cnt);
    k_final    <<<1,     128, 0, stream>>>(pool, cnt, agg1, rooti, fcW, fcb, out);
}
